// TriangleMultiplication_5789615915155
// MI455X (gfx1250) — matrix operand order repaired and checked by reading, not compiled, not run on hardware
//
#include <hip/hip_runtime.h>
#include <hip/hip_bf16.h>
#include <math.h>

// ---------------------------------------------------------------------------
// Triangle multiplication (outgoing), B=1 N=512 C=128, fp32 in/out.
// Strategy: f16 WMMA (v_wmma_f32_16x16x32_f16) for all GEMMs, f32 accum.
//   p1: LN(z) -> gated left/right (channel-major f16) + ogate (pos-major f16)
//   p2: per-channel 512x512x512 GEMM U_c = L_c * R_c^T  (128x128 tile / WG)
//       with double-buffered GLOBAL_LOAD_ASYNC_TO_LDS_B128 staging (ASYNCcnt)
//   p3: LN(U) @ W_out^T * og  -> f32 out
// Workspace: 6 f16 weight mats + left_t + right_t + og + update  (~256.3 MB)
// ---------------------------------------------------------------------------

typedef _Float16 f16;
typedef _Float16 h8   __attribute__((ext_vector_type(8)));
typedef _Float16 v16h __attribute__((ext_vector_type(16)));
typedef float    v8f  __attribute__((ext_vector_type(8)));

union V16 { v16h v; h8 h[2]; };

#define NN    512
#define CDIM  128
#define NPOS  (512 * 512)

static __device__ __forceinline__ v8f zero8() {
  v8f z = {0.f, 0.f, 0.f, 0.f, 0.f, 0.f, 0.f, 0.f};
  return z;
}

static __device__ __forceinline__ float sigm(float x) {
  return 1.0f / (1.0f + expf(-x));
}

// CDNA5 async memory->LDS copy (VGLOBAL GV mode): VDST = 32-bit LDS byte
// address (wave-relative, == low 32 bits of the flat shared pointer),
// VADDR = 64-bit global address. Tracked with ASYNCcnt.
static __device__ __forceinline__ void async_cp16(unsigned lds, const void* g) {
  asm volatile("global_load_async_to_lds_b128 %0, %1, off"
               :
               : "v"(lds), "v"(g)
               : "memory");
}

static __device__ __forceinline__ void wait_async0() {
  asm volatile("s_wait_asynccnt 0" ::: "memory");
}

static __device__ __forceinline__ unsigned lds_addr(const void* p) {
  return (unsigned)(uintptr_t)p;   // flat LDS address: offset in [31:0]
}

// One 16x16 output tile, K=128: 4 x wmma_f32_16x16x32_f16.
// A fragments prebuilt from LDS; B operand = f16 weight matrix, row-major
// [out_ch][in_ch] (torch Linear), so B[k][n] = W[n][k]: lane n reads a
// contiguous 32B run of row n -> k = 16h + s, same order as the A frags.
static __device__ __forceinline__ v8f mmtile128(const f16* __restrict__ W,
                                                int col, int kb2,
                                                const V16* afr) {
  v8f acc = zero8();
  const f16* wr = W + col * CDIM + kb2;
#pragma unroll
  for (int kb = 0; kb < 4; ++kb) {
    v16h b = *(const v16h*)(wr + kb * 32);   // 32B aligned: col*256 + {0,32} + kb*64
    acc = __builtin_amdgcn_wmma_f32_16x16x32_f16(false, afr[kb].v, false, b,
                                                 (short)0, acc, false, false);
  }
  return acc;
}

// ---------------------------------------------------------------------------
// Kernel 0: convert 6 weight matrices fp32 -> f16 into workspace.
// Order: [W_left, W_right, W_lgate, W_rgate, W_ogate, W_out]
// ---------------------------------------------------------------------------
__global__ __launch_bounds__(256) void k_prep(const float* __restrict__ wl,
                                              const float* __restrict__ wr,
                                              const float* __restrict__ wlg,
                                              const float* __restrict__ wrg,
                                              const float* __restrict__ wog,
                                              const float* __restrict__ wout,
                                              f16* __restrict__ dst) {
  int idx = blockIdx.x * 256 + threadIdx.x;   // 384 blocks * 256 = 98304 = 6*16384
  int m = idx >> 14;
  int off = idx & 16383;
  const float* s;
  switch (m) {
    case 0:  s = wl;   break;
    case 1:  s = wr;   break;
    case 2:  s = wlg;  break;
    case 3:  s = wrg;  break;
    case 4:  s = wog;  break;
    default: s = wout; break;
  }
  dst[idx] = (f16)s[off];
}

// ---------------------------------------------------------------------------
// Kernel 1: LN + 5 projections + gating. 64 positions / WG, 8 waves.
// ---------------------------------------------------------------------------
__global__ __launch_bounds__(256) void k_p1(const float* __restrict__ z,
                                            const float* __restrict__ mask,
                                            const float* __restrict__ lnw,
                                            const float* __restrict__ lnb,
                                            const float* __restrict__ blg,
                                            const float* __restrict__ brg,
                                            const float* __restrict__ bog,
                                            const f16* __restrict__ wf,
                                            f16* __restrict__ leftT,
                                            f16* __restrict__ rightT,
                                            f16* __restrict__ ogp) {
  __shared__ __align__(16) f16 sZn[64][136];   // pad 8: conflict-free 16B A-frag reads
  __shared__ __align__(16) f16 sL[128][72];    // [channel][pos], pad 8 (144B rows)
  __shared__ __align__(16) f16 sR[128][72];
  __shared__ float sLnW[128], sLnB[128], sMask[64];

  const int t = threadIdx.x;
  const int p0 = blockIdx.x * 64;              // 64 consecutive flat positions

  if (t < 128) { sLnW[t] = lnw[t]; sLnB[t] = lnb[t]; }
  if (t < 64)  { sMask[t] = mask[p0 + t]; }

  // ---- LayerNorm: 4 lanes per row, 32 channels each, shuffle reduce ----
  const int r = t >> 2, s4 = t & 3;
  const float* zrow = z + (size_t)(p0 + r) * CDIM + s4 * 32;
  float x[32];
#pragma unroll
  for (int q = 0; q < 8; ++q) {
    float4 v = ((const float4*)zrow)[q];
    x[4 * q + 0] = v.x; x[4 * q + 1] = v.y;
    x[4 * q + 2] = v.z; x[4 * q + 3] = v.w;
  }
  float sum = 0.f, sq = 0.f;
#pragma unroll
  for (int j = 0; j < 32; ++j) { sum += x[j]; sq += x[j] * x[j]; }
  sum += __shfl_xor(sum, 1); sum += __shfl_xor(sum, 2);
  sq  += __shfl_xor(sq, 1);  sq  += __shfl_xor(sq, 2);
  const float mu   = sum * (1.f / 128.f);
  const float var  = sq * (1.f / 128.f) - mu * mu;
  const float rstd = rsqrtf(var + 1e-5f);

  __syncthreads();                             // sLnW/sLnB/sMask ready
#pragma unroll
  for (int u = 0; u < 4; ++u) {
    h8 hv;
#pragma unroll
    for (int e = 0; e < 8; ++e) {
      int c = s4 * 32 + u * 8 + e;
      hv[e] = (f16)((x[u * 8 + e] - mu) * rstd * sLnW[c] + sLnB[c]);
    }
    *(h8*)&sZn[r][s4 * 32 + u * 8] = hv;
  }
  __syncthreads();

  // ---- GEMM: wave owns 16 rows (mb) and 4 column blocks (half) ----
  const int wave = t >> 5, lane = t & 31;
  const int mb = wave & 3, half = wave >> 2;
  const int m0 = mb * 16;
  const int mr = m0 + (lane & 15);
  const int ko = (lane >> 4) * 16;             // A-frag K half (contiguous 16)
  const int kb2 = (lane >> 4) * 16;            // B-frag K half (contiguous 16, as A)
  const int hi8 = (lane >> 4) * 8;             // C/D row offset for lanes 16-31

  V16 afr[4];
#pragma unroll
  for (int kb = 0; kb < 4; ++kb) {
    afr[kb].h[0] = *(const h8*)&sZn[mr][kb * 32 + ko];
    afr[kb].h[1] = *(const h8*)&sZn[mr][kb * 32 + ko + 8];
  }

  const f16* WL  = wf;
  const f16* WR  = wf + 16384;
  const f16* WLG = wf + 2 * 16384;
  const f16* WRG = wf + 3 * 16384;
  const f16* WOG = wf + 4 * 16384;

  for (int nb = half * 4; nb < half * 4 + 4; ++nb) {
    const int col = nb * 16 + (lane & 15);     // global output channel

    // left * sigmoid(lgate + b) * mask
    {
      v8f g = mmtile128(WLG, col, kb2, afr);
      v8f l = mmtile128(WL,  col, kb2, afr);
      float bl = blg[col];
#pragma unroll
      for (int r8 = 0; r8 < 8; ++r8) {
        int m = m0 + r8 + hi8;
        float val = l[r8] * sigm(g[r8] + bl) * sMask[m];
        sL[col][m] = (f16)val;
      }
    }
    // right * sigmoid(rgate + b) * mask
    {
      v8f g = mmtile128(WRG, col, kb2, afr);
      v8f rr = mmtile128(WR, col, kb2, afr);
      float br = brg[col];
#pragma unroll
      for (int r8 = 0; r8 < 8; ++r8) {
        int m = m0 + r8 + hi8;
        float val = rr[r8] * sigm(g[r8] + br) * sMask[m];
        sR[col][m] = (f16)val;
      }
    }
    // ogate: fold final mask in here too
    {
      v8f o = mmtile128(WOG, col, kb2, afr);
      float bo = bog[col];
#pragma unroll
      for (int r8 = 0; r8 < 8; ++r8) {
        int m = m0 + r8 + hi8;
        float val = sigm(o[r8] + bo) * sMask[m];
        ogp[(size_t)(p0 + m) * CDIM + col] = (f16)val;
      }
    }
  }
  __syncthreads();

  // ---- coalesced channel-major stores: left_t/right_t [c][pos] ----
  {
    const int c = t >> 1, part = t & 1;
    const h8* srcL = (const h8*)&sL[c][part * 32];
    const h8* srcR = (const h8*)&sR[c][part * 32];
    h8* dl = (h8*)(leftT  + (size_t)c * NPOS + p0 + part * 32);
    h8* dr = (h8*)(rightT + (size_t)c * NPOS + p0 + part * 32);
#pragma unroll
    for (int q = 0; q < 4; ++q) { dl[q] = srcL[q]; dr[q] = srcR[q]; }
  }
}

// ---------------------------------------------------------------------------
// Kernel 2: batched GEMM per channel: U_c = L_c * R_c^T, 128x128x512 tile/WG.
// grid = 128 channels * 16 tiles. Double-buffered async global->LDS staging:
// each k-step's 2x8KB tiles are DMA'd with global_load_async_to_lds_b128
// while the previous tiles feed 64 WMMAs.
// ---------------------------------------------------------------------------
__global__ __launch_bounds__(256) void k_p2(const f16* __restrict__ L,
                                            const f16* __restrict__ R,
                                            f16* __restrict__ U) {
  __shared__ __align__(16) f16 sA[2][128][40];   // [buf][i][k], pad 8 -> 80B rows
  __shared__ __align__(16) f16 sB[2][128][40];   // [buf][j][k]

  const int t = threadIdx.x;
  const int c  = blockIdx.x >> 4;
  const int i0 = ((blockIdx.x >> 2) & 3) * 128;
  const int j0 = (blockIdx.x & 3) * 128;
  const f16* Lb = L + (size_t)c * NPOS;
  const f16* Rb = R + (size_t)c * NPOS;

  const int wave = t >> 5, lane = t & 31;
  const int m0 = wave * 16;
  const int mr = m0 + (lane & 15);
  const int ko  = (lane >> 4) * 16;
  const int kb2 = (lane >> 4) * 16;
  const int rr = t >> 1, part = t & 1;           // staging: 2 threads per row

  // per-thread source rows (32B each per k-step) and LDS slots
  const f16* gA = Lb + (size_t)(i0 + rr) * NN + part * 16;
  const f16* gB = Rb + (size_t)(j0 + rr) * NN + part * 16;

  v8f acc[8];
#pragma unroll
  for (int jb = 0; jb < 8; ++jb) acc[jb] = zero8();

  // prologue: stage k-step 0 into buffer 0
  {
    unsigned la = lds_addr(&sA[0][rr][part * 16]);
    unsigned lb = lds_addr(&sB[0][rr][part * 16]);
    async_cp16(la,      gA);     async_cp16(la + 16, gA + 8);
    async_cp16(lb,      gB);     async_cp16(lb + 16, gB + 8);
  }

  for (int kk = 0; kk < 16; ++kk) {
    const int cur = kk & 1;
    wait_async0();               // this wave's async writes have landed in LDS
    __syncthreads();             // ... and everyone else's too; prev compute done

    if (kk < 15) {               // stage next k-step into the other buffer
      const int nxt = cur ^ 1;
      const f16* pa = gA + (kk + 1) * 32;
      const f16* pb = gB + (kk + 1) * 32;
      unsigned la = lds_addr(&sA[nxt][rr][part * 16]);
      unsigned lb = lds_addr(&sB[nxt][rr][part * 16]);
      async_cp16(la,      pa);   async_cp16(la + 16, pa + 8);
      async_cp16(lb,      pb);   async_cp16(lb + 16, pb + 8);
    }

    V16 a;
    a.h[0] = *(const h8*)&sA[cur][mr][ko];
    a.h[1] = *(const h8*)&sA[cur][mr][ko + 8];
#pragma unroll
    for (int jb = 0; jb < 8; ++jb) {
      V16 b;
      const int jr = jb * 16 + (lane & 15);
      b.h[0] = *(const h8*)&sB[cur][jr][kb2];
      b.h[1] = *(const h8*)&sB[cur][jr][kb2 + 8];
      acc[jb] = __builtin_amdgcn_wmma_f32_16x16x32_f16(false, a.v, false, b.v,
                                                       (short)0, acc[jb],
                                                       false, false);
    }
  }

  // epilogue: scale by 1/sqrt(N), store f16 channel-major [c][i][j]
  const float scale = 0.044194173824159216f;   // 1/sqrt(512)
  const int hi8 = (lane >> 4) * 8;
#pragma unroll
  for (int jb = 0; jb < 8; ++jb) {
#pragma unroll
    for (int r8 = 0; r8 < 8; ++r8) {
      int i = i0 + m0 + r8 + hi8;
      int j = j0 + jb * 16 + (lane & 15);
      U[(size_t)c * NPOS + (size_t)i * NN + j] = (f16)(acc[jb][r8] * scale);
    }
  }
}

// ---------------------------------------------------------------------------
// Kernel 3: LN(update) @ W_out^T * og -> f32 out. 64 positions / WG.
// ---------------------------------------------------------------------------
__global__ __launch_bounds__(256) void k_p3(const f16* __restrict__ U,
                                            const f16* __restrict__ OG,
                                            const float* __restrict__ onw,
                                            const float* __restrict__ onb,
                                            const f16* __restrict__ wf,
                                            float* __restrict__ out) {
  __shared__ __align__(16) f16 sU[64][136];    // [pos][c] transposed view of U
  __shared__ __align__(16) f16 sZn[64][136];
  __shared__ __align__(16) f16 sOg[64 * 128];
  __shared__ float sW[128], sBb[128];

  const int t = threadIdx.x;
  const int p0 = blockIdx.x * 64;

  if (t < 128) { sW[t] = onw[t]; sBb[t] = onb[t]; }

  // og tile: straight coalesced copy, layout matches global [pos][c]
  {
    const h8* so = (const h8*)(OG + (size_t)p0 * CDIM) + t * 4;
    h8* dsto = (h8*)sOg + t * 4;
#pragma unroll
    for (int q = 0; q < 4; ++q) dsto[q] = so[q];
  }
  // update: coalesced channel-major reads, transpose-scatter into LDS [pos][c]
  {
    const int cc = t >> 1, part = t & 1;
    const h8* su = (const h8*)(U + (size_t)cc * NPOS + p0 + part * 32);
#pragma unroll
    for (int q = 0; q < 4; ++q) {
      h8 hv = su[q];
#pragma unroll
      for (int e = 0; e < 8; ++e) sU[part * 32 + q * 8 + e][cc] = hv[e];
    }
  }
  __syncthreads();

  // ---- LayerNorm over channels ----
  const int r = t >> 2, s4 = t & 3;
  float x[32];
#pragma unroll
  for (int j = 0; j < 32; ++j) x[j] = (float)sU[r][s4 * 32 + j];
  float sum = 0.f, sq = 0.f;
#pragma unroll
  for (int j = 0; j < 32; ++j) { sum += x[j]; sq += x[j] * x[j]; }
  sum += __shfl_xor(sum, 1); sum += __shfl_xor(sum, 2);
  sq  += __shfl_xor(sq, 1);  sq  += __shfl_xor(sq, 2);
  const float mu   = sum * (1.f / 128.f);
  const float var  = sq * (1.f / 128.f) - mu * mu;
  const float rstd = rsqrtf(var + 1e-5f);
#pragma unroll
  for (int u = 0; u < 4; ++u) {
    h8 hv;
#pragma unroll
    for (int e = 0; e < 8; ++e) {
      int c = s4 * 32 + u * 8 + e;
      hv[e] = (f16)((x[u * 8 + e] - mu) * rstd * sW[c] + sBb[c]);
    }
    *(h8*)&sZn[r][s4 * 32 + u * 8] = hv;
  }
  __syncthreads();

  // ---- GEMM with W_out, fused og multiply, f32 store ----
  const int wave = t >> 5, lane = t & 31;
  const int mb = wave & 3, half = wave >> 2;
  const int m0 = mb * 16;
  const int mr = m0 + (lane & 15);
  const int ko  = (lane >> 4) * 16;
  const int kb2 = (lane >> 4) * 16;
  const int hi8 = (lane >> 4) * 8;

  V16 afr[4];
#pragma unroll
  for (int kb = 0; kb < 4; ++kb) {
    afr[kb].h[0] = *(const h8*)&sZn[mr][kb * 32 + ko];
    afr[kb].h[1] = *(const h8*)&sZn[mr][kb * 32 + ko + 8];
  }

  const f16* WOUT = wf + 5 * 16384;
  for (int nb = half * 4; nb < half * 4 + 4; ++nb) {
    const int col = nb * 16 + (lane & 15);
    v8f acc = mmtile128(WOUT, col, kb2, afr);
#pragma unroll
    for (int r8 = 0; r8 < 8; ++r8) {
      int m = m0 + r8 + hi8;
      float val = acc[r8] * (float)sOg[m * CDIM + col];
      out[(size_t)(p0 + m) * CDIM + col] = val;
    }
  }
}

// ---------------------------------------------------------------------------
// Launch
// ---------------------------------------------------------------------------
extern "C" void kernel_launch(void* const* d_in, const int* in_sizes, int n_in,
                              void* d_out, int out_size, void* d_ws, size_t ws_size,
                              hipStream_t stream) {
  const float* z        = (const float*)d_in[0];
  const float* mask     = (const float*)d_in[1];
  const float* ln_w     = (const float*)d_in[2];
  const float* ln_b     = (const float*)d_in[3];
  const float* W_left   = (const float*)d_in[4];
  const float* W_right  = (const float*)d_in[5];
  const float* W_lgate  = (const float*)d_in[6];
  const float* b_lgate  = (const float*)d_in[7];
  const float* W_rgate  = (const float*)d_in[8];
  const float* b_rgate  = (const float*)d_in[9];
  const float* on_w     = (const float*)d_in[10];
  const float* on_b     = (const float*)d_in[11];
  const float* W_out    = (const float*)d_in[12];
  const float* W_ogate  = (const float*)d_in[13];
  const float* b_ogate  = (const float*)d_in[14];
  (void)in_sizes; (void)n_in; (void)out_size;

  // workspace layout (bytes): weights(256K) | leftT | rightT | og | update
  const size_t SZ_MAT = (size_t)128 * NPOS * sizeof(f16);  // 64 MiB each
  char* ws = (char*)d_ws;
  f16* wf16   = (f16*)ws;
  f16* leftT  = (f16*)(ws + 262144);
  f16* rightT = (f16*)(ws + 262144 + SZ_MAT);
  f16* og     = (f16*)(ws + 262144 + 2 * SZ_MAT);
  f16* upd    = (f16*)(ws + 262144 + 3 * SZ_MAT);
  (void)ws_size;  // requires ~256.3 MB

  k_prep<<<384, 256, 0, stream>>>(W_left, W_right, W_lgate, W_rgate, W_ogate,
                                  W_out, wf16);
  k_p1<<<NPOS / 64, 256, 0, stream>>>(z, mask, ln_w, ln_b, b_lgate, b_rgate,
                                      b_ogate, wf16, leftT, rightT, og);
  k_p2<<<128 * 16, 256, 0, stream>>>(leftT, rightT, upd);
  k_p3<<<NPOS / 64, 256, 0, stream>>>(upd, og, on_w, on_b, wf16,
                                      (float*)d_out);
}
